// TemporalSpectralFilter_1116691497519
// MI455X (gfx1250) — matrix operand order repaired and checked by reading, not compiled, not run on hardware
//
#include <hip/hip_runtime.h>
#include <hip/hip_bf16.h>

typedef __attribute__((ext_vector_type(16))) _Float16 v16h;
typedef __attribute__((ext_vector_type(8)))  float    v8f;
typedef __attribute__((ext_vector_type(4)))  unsigned int u32x4;
typedef __attribute__((ext_vector_type(8)))  int      i32x8;
typedef __attribute__((ext_vector_type(4)))  int      i32x4;

#define U_NODES 100000
#define I_NODES 50000
#define N_NODES 150000
#define E_EDGES 1000000
#define K_BANDS 4
#define DF      128
#define DLAT    64
#define HIDDEN  64
#define TEMP    0.7f
#define SLOPE   0.01f

#define SA_PITCH 132   // 128 + 4 (TDM pad_amount) -> conflict-free column reads
#define SH_PITCH 264   // 256 + 8

#if __has_builtin(__builtin_amdgcn_tensor_load_to_lds)
#define USE_TDM 1
#else
#define USE_TDM 0
#endif

// ---------------------------------------------------------------------------
// Pack fp32 weight matrix into f16 WMMA B-fragment order:
// wp[((nt*nkk + kk)*32 + lane)*16 + j] = (f16) w[(kk*32 + (j>>3)*16 + (lane>>4)*8 + (j&7))*ncols + nt*16 + (lane&15)]
// so each lane's 16 halves for one WMMA step are 32 contiguous bytes.
// ---------------------------------------------------------------------------
__global__ __launch_bounds__(256)
void pack_w_kernel(const float* __restrict__ w, _Float16* __restrict__ wp,
                   int ncols, int nkk, int total)
{
    int idx = blockIdx.x * 256 + threadIdx.x;
    if (idx >= total) return;
    int j    = idx & 15;
    int lane = (idx >> 4) & 31;
    int kk   = (idx >> 9) % nkk;
    int nt   = (idx >> 9) / nkk;
    int col  = nt * 16 + (lane & 15);
    int row  = kk * 32 + (j >> 3) * 16 + (lane >> 4) * 8 + (j & 7);
    wp[idx] = (_Float16)w[(size_t)row * ncols + col];
}

// ---------------------------------------------------------------------------
// Feature MLP: proj = leaky_relu(features @ W1 + b1) @ W2 + b2
// One block (8 wave32) per 16-row tile; TDM loads the feature tile into LDS
// (with hardware padding to SA_PITCH); WMMA f32 <- f16 x f16.
// ---------------------------------------------------------------------------
__global__ __launch_bounds__(256)
void feat_mlp_kernel(const float* __restrict__ features,
                     const _Float16* __restrict__ wp1, const float* __restrict__ b1,
                     const _Float16* __restrict__ wp2, const float* __restrict__ b2,
                     float* __restrict__ node_init)
{
    __shared__ float smem[16 * SA_PITCH + 16 * SH_PITCH];
    float* sA = smem;                       // 16 x 128 feature tile (pitch 132), LDS offset 0
    float* sH = smem + 16 * SA_PITCH;       // 16 x 256 hidden tile  (pitch 264)

    const int tid  = threadIdx.x;
    const int lane = tid & 31;
    const int wave = tid >> 5;
    const int row0 = blockIdx.x * 16;

#if USE_TDM
    if (wave == 0) {
        // Tensor DMA: 2D tile 128x16 f32, global row stride 128, LDS pad 4 DWORDs
        // after every 128 DWORDs -> row pitch 132 DWORDs (ISA 8.3/8.4 D# layout).
        unsigned long long ga =
            (unsigned long long)(uintptr_t)(const void*)(features + (size_t)row0 * DF);
        u32x4 g0;
        g0[0] = 1u;                                        // count=1, user descriptor
        g0[1] = 0u;                                        // lds_addr: sA at LDS offset 0
        g0[2] = (unsigned)(ga & 0xffffffffu);              // global_addr[31:0]
        g0[3] = (unsigned)((ga >> 32) & 0x01ffffffu)       // global_addr[56:32]
              | (2u << 30);                                // type = 2 ("image")
        i32x8 g1;
        g1[0] = (2 << 16) | (1 << 20) | (6 << 22) | (3 << 25); // 4B elems, pad_en,
                                                               // interval=128dw, amount=4dw
        g1[1] = (int)(DF << 16);                           // tensor_dim0 = 128 (lo16)
        g1[2] = (int)(16u << 16);                          // tensor_dim1 = 16  (lo16)
        g1[3] = (int)(DF << 16);                           // tile_dim0   = 128
        g1[4] = 16;                                        // tile_dim1=16, tile_dim2=0
        g1[5] = DF;                                        // tensor_dim0_stride = 128
        g1[6] = 0;
        g1[7] = 0;
        i32x4 z4 = {0, 0, 0, 0};
#if __clang_major__ >= 23
        i32x8 z8 = {0, 0, 0, 0, 0, 0, 0, 0};
        __builtin_amdgcn_tensor_load_to_lds(g0, g1, z4, z4, z8, 0);
#else
        __builtin_amdgcn_tensor_load_to_lds(g0, g1, z4, z4, 0);
#endif
        __builtin_amdgcn_s_wait_tensorcnt(0);
    }
#else
    for (int i = tid; i < 16 * DF; i += 256) {
        int r = i >> 7, c = i & 127;
        sA[r * SA_PITCH + c] = features[(size_t)(row0 + r) * DF + c];
    }
#endif
    __syncthreads();

    const int m    = lane & 15;   // row (A) or col (B/C) index within tile
    const int half = lane >> 4;   // lane group 0/1

    // ---- stage 1: hidden[16,256] ----
    for (int nt = wave * 2; nt < wave * 2 + 2; ++nt) {
        const int col = nt * 16 + m;
        v8f acc = {};
        for (int kk = 0; kk < DF / 32; ++kk) {
            const int kb = kk * 32;
            v16h a;
#pragma unroll
            for (int j = 0; j < 8; ++j) {
                a[j]     = (_Float16)sA[m * SA_PITCH + kb + half * 8 + j];
                a[8 + j] = (_Float16)sA[m * SA_PITCH + kb + 16 + half * 8 + j];
            }
            v16h b = *(const v16h*)&wp1[(size_t)(((nt * 4 + kk) * 32 + lane)) * 16];
            acc = __builtin_amdgcn_wmma_f32_16x16x32_f16(false, a, false, b,
                                                         (short)0, acc, false, false);
        }
#pragma unroll
        for (int r = 0; r < 8; ++r) {
            int mm = r + half * 8;
            float v = acc[r] + b1[col];
            v = (v >= 0.f) ? v : v * SLOPE;               // leaky_relu(0.01)
            sH[mm * SH_PITCH + nt * 16 + m] = v;
        }
    }
    __syncthreads();

    // ---- stage 2: out[16,64] = hidden @ W2 + b2 ----
    if (wave < 4) {
        const int col = wave * 16 + m;
        v8f acc = {};
        for (int kk = 0; kk < 256 / 32; ++kk) {
            const int kb = kk * 32;
            v16h a;
#pragma unroll
            for (int j = 0; j < 8; ++j) {
                a[j]     = (_Float16)sH[m * SH_PITCH + kb + half * 8 + j];
                a[8 + j] = (_Float16)sH[m * SH_PITCH + kb + 16 + half * 8 + j];
            }
            v16h b = *(const v16h*)&wp2[(size_t)(((wave * 8 + kk) * 32 + lane)) * 16];
            acc = __builtin_amdgcn_wmma_f32_16x16x32_f16(false, a, false, b,
                                                         (short)0, acc, false, false);
        }
#pragma unroll
        for (int r = 0; r < 8; ++r) {
            int mm = r + half * 8;
            node_init[(size_t)(U_NODES + row0 + mm) * DLAT + col] = acc[r] + b2[col];
        }
    }
}

// ---------------------------------------------------------------------------
__global__ __launch_bounds__(256)
void pref_copy_kernel(const float* __restrict__ pref,
                      float* __restrict__ node_init,
                      float* __restrict__ out_pref, size_t n4)
{
    size_t i = (size_t)blockIdx.x * blockDim.x + threadIdx.x;
    if (i < n4) {
        float4 v = ((const float4*)pref)[i];
        ((float4*)node_init)[i] = v;
        ((float4*)out_pref)[i]  = v;
    }
}

// one wave32 per 64-wide row, shuffle reduction
__global__ __launch_bounds__(256)
void normalize_kernel(float* __restrict__ x)
{
    int wave = threadIdx.x >> 5;
    int lane = threadIdx.x & 31;
    int row  = blockIdx.x * 8 + wave;
    if (row >= N_NODES) return;
    float a = x[(size_t)row * DLAT + lane];
    float b = x[(size_t)row * DLAT + 32 + lane];
    float ss = a * a + b * b;
#pragma unroll
    for (int off = 16; off > 0; off >>= 1)
        ss += __shfl_xor(ss, off, 32);
    float inv = 1.0f / fmaxf(sqrtf(ss), 1e-12f);
    x[(size_t)row * DLAT + lane]      = a * inv;
    x[(size_t)row * DLAT + 32 + lane] = b * inv;
}

__global__ __launch_bounds__(256)
void zero_kernel(float* __restrict__ p, size_t n4)
{
    size_t i = (size_t)blockIdx.x * blockDim.x + threadIdx.x;
    if (i < n4) ((float4*)p)[i] = make_float4(0.f, 0.f, 0.f, 0.f);
}

__global__ __launch_bounds__(256)
void degree_kernel(const int* __restrict__ src, const float* __restrict__ w,
                   float* __restrict__ deg)
{
    int e = blockIdx.x * 256 + threadIdx.x;
    int k = blockIdx.y;
    if (e < E_EDGES)
        atomicAdd(&deg[(size_t)k * N_NODES + src[e]], w[(size_t)k * E_EDGES + e]);
}

__global__ __launch_bounds__(256)
void dinv_kernel(float* __restrict__ deg, size_t n)
{
    size_t i = (size_t)blockIdx.x * blockDim.x + threadIdx.x;
    if (i < n) deg[i] = 1.0f / sqrtf(fmaxf(deg[i], 1e-12f));
}

__global__ __launch_bounds__(256)
void coef_kernel(const int* __restrict__ src, const int* __restrict__ dst,
                 const float* __restrict__ w, const float* __restrict__ dinv,
                 float* __restrict__ coef, int k)
{
    int e = blockIdx.x * 256 + threadIdx.x;
    if (e < E_EDGES)
        coef[e] = dinv[(size_t)k * N_NODES + src[e]] * w[(size_t)k * E_EDGES + e]
                * dinv[(size_t)k * N_NODES + dst[e]];
}

// 16 threads per edge, each moves a float4 of the 64-dim row; L2 atomics
__global__ __launch_bounds__(256)
void scatter_kernel(const int* __restrict__ src, const int* __restrict__ dst,
                    const float* __restrict__ coef,
                    const float* __restrict__ x, float* __restrict__ out)
{
    size_t t = (size_t)blockIdx.x * 256 + threadIdx.x;
    int e = (int)(t >> 4);
    int q = ((int)t & 15) * 4;
    if (e < E_EDGES) {
        float c = coef[e];
        int s = src[e], d = dst[e];
        float4 v = *(const float4*)&x[(size_t)s * DLAT + q];
        float* o = &out[(size_t)d * DLAT + q];
        atomicAdd(o + 0, c * v.x);
        atomicAdd(o + 1, c * v.y);
        atomicAdd(o + 2, c * v.z);
        atomicAdd(o + 3, c * v.w);
    }
}

__global__ __launch_bounds__(256)
void add3_kernel(const float* __restrict__ x, const float* __restrict__ h2,
                 float* __restrict__ hat, size_t n4)   // hat = x + hat + h2
{
    size_t i = (size_t)blockIdx.x * blockDim.x + threadIdx.x;
    if (i < n4) {
        float4 a = ((const float4*)x)[i];
        float4 b = ((float4*)hat)[i];
        float4 c = ((const float4*)h2)[i];
        ((float4*)hat)[i] = make_float4(a.x + b.x + c.x, a.y + b.y + c.y,
                                        a.z + b.z + c.z, a.w + b.w + c.w);
    }
}

__global__ __launch_bounds__(256)
void gate_kernel(const float* __restrict__ user_state, const float* __restrict__ item_state,
                 const float* __restrict__ w_u, const float* __restrict__ b_u,
                 const float* __restrict__ w_i, const float* __restrict__ b_i,
                 const float* __restrict__ w_head, const float* __restrict__ b_head,
                 float* __restrict__ gate)
{
    __shared__ float swu[7 * HIDDEN], swi[7 * HIDDEN], swh[HIDDEN * K_BANDS];
    __shared__ float sbu[HIDDEN], sbi[HIDDEN], sbh[K_BANDS];
    int tid = threadIdx.x;
    for (int i = tid; i < 7 * HIDDEN; i += 256) { swu[i] = w_u[i]; swi[i] = w_i[i]; }
    for (int i = tid; i < HIDDEN * K_BANDS; i += 256) swh[i] = w_head[i];
    for (int i = tid; i < HIDDEN; i += 256) { sbu[i] = b_u[i]; sbi[i] = b_i[i]; }
    if (tid < K_BANDS) sbh[tid] = b_head[tid];
    __syncthreads();

    int n = blockIdx.x * 256 + tid;
    if (n >= N_NODES) return;
    float s[7];
    const float *w, *bb;
    if (n < U_NODES) {
        for (int j = 0; j < 7; ++j) s[j] = user_state[(size_t)n * 7 + j];
        w = swu; bb = sbu;
    } else {
        int m = n - U_NODES;
        for (int j = 0; j < 7; ++j) s[j] = item_state[(size_t)m * 7 + j];
        w = swi; bb = sbi;
    }
    float lg[K_BANDS] = {sbh[0], sbh[1], sbh[2], sbh[3]};
    for (int h = 0; h < HIDDEN; ++h) {
        float acc = bb[h];
#pragma unroll
        for (int j = 0; j < 7; ++j) acc += s[j] * w[j * HIDDEN + h];
        acc = fmaxf(acc, 0.f);
#pragma unroll
        for (int k = 0; k < K_BANDS; ++k) lg[k] += acc * swh[h * K_BANDS + k];
    }
    float mx = fmaxf(fmaxf(lg[0], lg[1]), fmaxf(lg[2], lg[3]));
    float ex[K_BANDS], ssum = 0.f;
#pragma unroll
    for (int k = 0; k < K_BANDS; ++k) { ex[k] = expf((lg[k] - mx) / TEMP); ssum += ex[k]; }
#pragma unroll
    for (int k = 0; k < K_BANDS; ++k) gate[(size_t)n * K_BANDS + k] = ex[k] / ssum;
}

__global__ __launch_bounds__(256)
void combine_kernel(const float* __restrict__ hat,   // [K,N,DL]
                    const float* __restrict__ gate,  // [N,K]
                    float* __restrict__ fe, size_t n4)
{
    size_t t = (size_t)blockIdx.x * blockDim.x + threadIdx.x;
    if (t >= n4) return;
    int n = (int)(t / (DLAT / 4));
    float4 acc = make_float4(0.f, 0.f, 0.f, 0.f);
#pragma unroll
    for (int k = 0; k < K_BANDS; ++k) {
        float g = gate[(size_t)n * K_BANDS + k];
        float4 v = ((const float4*)hat)[(size_t)k * N_NODES * (DLAT / 4) + t];
        acc.x += g * v.x; acc.y += g * v.y; acc.z += g * v.z; acc.w += g * v.w;
    }
    ((float4*)fe)[t] = acc;
}

// ---------------------------------------------------------------------------
extern "C" void kernel_launch(void* const* d_in, const int* in_sizes, int n_in,
                              void* d_out, int out_size, void* d_ws, size_t ws_size,
                              hipStream_t stream)
{
    (void)in_sizes; (void)n_in; (void)out_size; (void)ws_size;
    const int*   edge_index = (const int*)  d_in[1];
    const float* features   = (const float*)d_in[2];
    const float* edge_w     = (const float*)d_in[3];
    const float* user_state = (const float*)d_in[4];
    const float* item_state = (const float*)d_in[5];
    const float* preference = (const float*)d_in[6];
    const float* w_feat1    = (const float*)d_in[7];
    const float* b_feat1    = (const float*)d_in[8];
    const float* w_feat2    = (const float*)d_in[9];
    const float* b_feat2    = (const float*)d_in[10];
    const float* w_u        = (const float*)d_in[11];
    const float* b_u        = (const float*)d_in[12];
    const float* w_i        = (const float*)d_in[13];
    const float* b_i        = (const float*)d_in[14];
    const float* w_head     = (const float*)d_in[15];
    const float* b_head     = (const float*)d_in[16];

    const int* src = edge_index;
    const int* dst = edge_index + E_EDGES;

    float* out_fe   = (float*)d_out;                              // [N, DL]
    float* out_hat  = out_fe  + (size_t)N_NODES * DLAT;           // [K, N, DL]
    float* out_pref = out_hat + (size_t)K_BANDS * N_NODES * DLAT; // [U, DL]

    float* ws        = (float*)d_ws;
    float* node_init = ws;                                        // N*DL
    float* tmp       = node_init + (size_t)N_NODES * DLAT;        // N*DL
    float* dinv      = tmp + (size_t)N_NODES * DLAT;              // K*N
    float* gate      = dinv + (size_t)K_BANDS * N_NODES;          // N*K
    float* coef      = gate + (size_t)N_NODES * K_BANDS;          // E
    _Float16* wp1    = (_Float16*)(coef + E_EDGES);               // 32768 halves (32B-aligned)
    _Float16* wp2    = wp1 + 32768;                               // 16384 halves

    // 0) pack weights into f16 WMMA B-fragment order (L2-resident, reused 3125x)
    pack_w_kernel<<<32768 / 256, 256, 0, stream>>>(w_feat1, wp1, 256, 4, 32768);
    pack_w_kernel<<<16384 / 256, 256, 0, stream>>>(w_feat2, wp2, 64, 8, 16384);

    // 1) feature MLP (TDM tile load + WMMA) -> item rows of node_init
    feat_mlp_kernel<<<I_NODES / 16, 256, 0, stream>>>(features, wp1, b_feat1,
                                                      wp2, b_feat2, node_init);
    // 2) preference -> user rows of node_init + output passthrough
    {
        size_t n4 = (size_t)U_NODES * DLAT / 4;
        pref_copy_kernel<<<(unsigned)((n4 + 255) / 256), 256, 0, stream>>>(
            preference, node_init, out_pref, n4);
    }
    // 3) F.normalize rows
    normalize_kernel<<<N_NODES / 8, 256, 0, stream>>>(node_init);
    // 4) weighted degrees -> dinv
    {
        size_t n4 = (size_t)K_BANDS * N_NODES / 4;
        zero_kernel<<<(unsigned)((n4 + 255) / 256), 256, 0, stream>>>(dinv, n4);
        dim3 g((E_EDGES + 255) / 256, K_BANDS);
        degree_kernel<<<g, 256, 0, stream>>>(src, edge_w, dinv);
        size_t n = (size_t)K_BANDS * N_NODES;
        dinv_kernel<<<(unsigned)((n + 255) / 256), 256, 0, stream>>>(dinv, n);
    }
    // 5) time-scale gate
    gate_kernel<<<(N_NODES + 255) / 256, 256, 0, stream>>>(
        user_state, item_state, w_u, b_u, w_i, b_i, w_head, b_head, gate);
    // 6) K spectral scales, 2 GCN hops each, out = x + h1 + h2
    size_t nd4 = (size_t)N_NODES * DLAT / 4;
    unsigned gz = (unsigned)((nd4 + 255) / 256);
    unsigned gs = (unsigned)((size_t)E_EDGES * 16 / 256);
    for (int k = 0; k < K_BANDS; ++k) {
        float* hatk = out_hat + (size_t)k * N_NODES * DLAT;
        coef_kernel<<<(E_EDGES + 255) / 256, 256, 0, stream>>>(src, dst, edge_w, dinv, coef, k);
        zero_kernel<<<gz, 256, 0, stream>>>(hatk, nd4);
        scatter_kernel<<<gs, 256, 0, stream>>>(src, dst, coef, node_init, hatk); // h1
        zero_kernel<<<gz, 256, 0, stream>>>(tmp, nd4);
        scatter_kernel<<<gs, 256, 0, stream>>>(src, dst, coef, hatk, tmp);       // h2
        add3_kernel<<<gz, 256, 0, stream>>>(node_init, tmp, hatk, nd4);
    }
    // 7) gated combine -> filter_emb
    combine_kernel<<<gz, 256, 0, stream>>>(out_hat, gate, out_fe, nd4);
}
